// SSLModel_71433896067588
// MI455X (gfx1250) — matrix operand order repaired and checked by reading, not compiled, not run on hardware
//
#include <hip/hip_runtime.h>
#include <hip/hip_bf16.h>

// ---------------------------------------------------------------------------
// Types for CDNA5 WMMA / async-to-LDS
// ---------------------------------------------------------------------------
typedef __bf16 bf16_t;
typedef __attribute__((ext_vector_type(16))) __bf16 v16bf;
typedef __attribute__((ext_vector_type(8)))  float  v8f;
typedef int v4i_vec __attribute__((vector_size(16)));   // b128 payload type

#define AS_GLOBAL __attribute__((address_space(1)))
#define AS_LDS    __attribute__((address_space(3)))

#if __has_builtin(__builtin_amdgcn_global_load_async_to_lds_b128) && \
    __has_builtin(__builtin_amdgcn_s_wait_asynccnt)
#define USE_ASYNC_LDS 1
#else
#define USE_ASYNC_LDS 0
#endif

#define NN 10000
#define EE 160000
#define GG 128

// ---------------------------------------------------------------------------
// Elementwise helpers (vectorized where sizes are multiples of 4)
// ---------------------------------------------------------------------------
__global__ void k_fill4(float4* __restrict__ p, float v, int n4) {
    int i = blockIdx.x * 256 + threadIdx.x;
    if (i < n4) p[i] = make_float4(v, v, v, v);
}

__global__ void k_cvt_bf16_v4(const float4* __restrict__ in, bf16_t* __restrict__ out, int n4) {
    int i = blockIdx.x * 256 + threadIdx.x;
    if (i < n4) {
        float4 v = in[i];
        union { bf16_t h[4]; uint2 u; } o;
        o.h[0] = (bf16_t)v.x; o.h[1] = (bf16_t)v.y;
        o.h[2] = (bf16_t)v.z; o.h[3] = (bf16_t)v.w;
        *(uint2*)(out + (unsigned)i * 4u) = o.u;
    }
}

__global__ void k_deg(const int* __restrict__ dst, float* __restrict__ deg, int e) {
    int i = blockIdx.x * 256 + threadIdx.x;
    if (i < e) atomicAdd(&deg[dst[i]], 1.0f);
}

__global__ void k_rsqrt(float* __restrict__ p, int n) {
    int i = blockIdx.x * 256 + threadIdx.x;
    if (i < n) p[i] = rsqrtf(p[i]);
}

__global__ void k_add(float* __restrict__ a, const float* __restrict__ b, int n) {
    int i = blockIdx.x * 256 + threadIdx.x;
    if (i < n) a[i] += b[i];
}

__global__ void k_sigmoid(const float* __restrict__ z, float* __restrict__ out, int n) {
    int i = blockIdx.x * 256 + threadIdx.x;
    if (i < n) out[i] = 1.0f / (1.0f + __expf(-z[i]));
}

// ---------------------------------------------------------------------------
// bf16 WMMA GEMM: C[M,N] = A[M,K] @ B[K,N] (+bias, +relu)
// Block tile 64x64, 4 waves (128 threads), K-step 32.
// A tile staged global->LDS with global_load_async_to_lds_b128 (ASYNCcnt);
// B tile staged transposed so both WMMA fragments are 2x b128 LDS reads.
// K is always a multiple of 32 here; M/N edges are guarded.
// ---------------------------------------------------------------------------
__global__ __launch_bounds__(128) void k_gemm_bf16(
    const bf16_t* __restrict__ A, const bf16_t* __restrict__ B,
    float* __restrict__ C, const float* __restrict__ bias,
    int M, int K, int Nn, int relu)
{
    __shared__ bf16_t sA[64][32];   // [m][k]
    __shared__ bf16_t sB[64][32];   // transposed: [n][k]

    const int tid  = threadIdx.x;
    const int wave = tid >> 5;
    const int lane = tid & 31;
    const int half = lane >> 4;     // 0: lanes 0-15, 1: lanes 16-31
    const int l16  = lane & 15;
    const int bm   = blockIdx.x * 64;
    const int bn   = blockIdx.y * 64;
    const bool nfull = ((Nn & 63) == 0);  // all conv GEMMs; head2 (N=1317) is not

    // A staging coords: thread t -> row t/2, 16 bf16 at col (t&1)*16
    const int ar = tid >> 1;
    const int ac = (tid & 1) << 4;
    const int agr = bm + ar;

    v8f acc[4] = {};

    for (int k0 = 0; k0 < K; k0 += 32) {
        __syncthreads();
        // ---- stage A tile (64x32 bf16)
        {
            const bf16_t* gA = A + (unsigned)agr * (unsigned)K + (unsigned)(k0 + ac);
#if USE_ASYNC_LDS
            if (agr < M) {
                __builtin_amdgcn_global_load_async_to_lds_b128(
                    (AS_GLOBAL v4i_vec*)gA, (AS_LDS v4i_vec*)&sA[ar][ac], 0, 0);
                __builtin_amdgcn_global_load_async_to_lds_b128(
                    (AS_GLOBAL v4i_vec*)(gA + 8), (AS_LDS v4i_vec*)&sA[ar][ac + 8], 0, 0);
            } else {
                uint4 zz = {0, 0, 0, 0};
                uint4* q = (uint4*)&sA[ar][ac];
                q[0] = zz; q[1] = zz;
            }
#else
            uint4 lo = {0, 0, 0, 0}, hi = {0, 0, 0, 0};
            if (agr < M) {
                const uint4* p = (const uint4*)gA;
                lo = p[0]; hi = p[1];
            }
            uint4* q = (uint4*)&sA[ar][ac];
            q[0] = lo; q[1] = hi;
#endif
        }
        // ---- stage B tile transposed (sB[n][k])
        if (nfull) {
            // full tile: each thread reads 8 contiguous bf16 (b128), twice
            #pragma unroll
            for (int p = 0; p < 2; ++p) {
                int q = p * 1024 + tid * 8;
                int k = q >> 6, n = q & 63;
                union { uint4 u; bf16_t e[8]; } bv;
                bv.u = *(const uint4*)(B + (unsigned)(k0 + k) * (unsigned)Nn + (unsigned)(bn + n));
                #pragma unroll
                for (int t = 0; t < 8; ++t) sB[n + t][k] = bv.e[t];
            }
        } else {
            #pragma unroll
            for (int j = 0; j < 16; ++j) {
                int q = j * 128 + tid;
                int k = q >> 6, n = q & 63;
                bf16_t v = (bf16_t)0.0f;
                if (bn + n < Nn) v = B[(unsigned)(k0 + k) * (unsigned)Nn + (unsigned)(bn + n)];
                sB[n][k] = v;
            }
        }
#if USE_ASYNC_LDS
        __builtin_amdgcn_s_wait_asynccnt(0);
#endif
        __syncthreads();

        // ---- A fragment: row = wave*16 + l16; K = {8h..8h+7, 16+8h..+7}
        union { uint4 u[2]; v16bf v; } af;
        {
            int mr = (wave << 4) | l16;
            af.u[0] = *(const uint4*)&sA[mr][8 * half];
            af.u[1] = *(const uint4*)&sA[mr][16 + 8 * half];
        }
        // ---- 4 B fragments, 4 WMMAs (wave covers full 64 output columns)
        #pragma unroll
        for (int jt = 0; jt < 4; ++jt) {
            union { uint4 u[2]; v16bf v; } bfg;
            int nr = (jt << 4) | l16;
            bfg.u[0] = *(const uint4*)&sB[nr][8 * half];
            bfg.u[1] = *(const uint4*)&sB[nr][16 + 8 * half];
            acc[jt] = __builtin_amdgcn_wmma_f32_16x16x32_bf16(
                false, af.v, false, bfg.v, (short)0, acc[jt], false, false);
        }
    }

    // ---- epilogue/store per C layout: VGPR r -> M = r + 8*half, N = l16
    #pragma unroll
    for (int jt = 0; jt < 4; ++jt) {
        int col = bn + (jt << 4) + l16;
        if (col >= Nn) continue;
        float bv = bias ? bias[col] : 0.0f;
        #pragma unroll
        for (int r = 0; r < 8; ++r) {
            int row = bm + (wave << 4) + (half << 3) + r;
            if (row < M) {
                float v = acc[jt][r] + bv;
                if (relu) v = fmaxf(v, 0.0f);
                C[(unsigned)row * (unsigned)Nn + (unsigned)col] = v;
            }
        }
    }
}

// ---------------------------------------------------------------------------
// Edge aggregation: z[dst] += dinv[src]*dinv[dst] * h[src]
// One block per (edge, 1024-feature chunk); float4 per lane; h/z are
// 80 MB each -> resident in the 192 MB L2, so atomics stay on-chip.
// ---------------------------------------------------------------------------
__global__ __launch_bounds__(256) void k_edge_agg(
    const float* __restrict__ h, const int* __restrict__ src,
    const int* __restrict__ dst, const float* __restrict__ dinv,
    float* __restrict__ z, int dim)
{
    int e = blockIdx.x;
    int s = src[e];
    int d = dst[e];
    float nrm = dinv[s] * dinv[d];
    unsigned f = (blockIdx.y << 10) + (threadIdx.x << 2);
    if (f < (unsigned)dim) {
        const float4 hv = *(const float4*)(h + (unsigned)s * (unsigned)dim + f);
        float* zp = z + (unsigned)d * (unsigned)dim + f;
        atomicAdd(zp + 0, nrm * hv.x);
        atomicAdd(zp + 1, nrm * hv.y);
        atomicAdd(zp + 2, nrm * hv.z);
        atomicAdd(zp + 3, nrm * hv.w);
    }
}

// self-loop term + bias (+relu), in place on z. dim = 1<<shift.
__global__ void k_conv_epi(float* __restrict__ z, const float* __restrict__ h,
                           const float* __restrict__ dinv, const float* __restrict__ bias,
                           int n, int shift, int relu)
{
    int idx = blockIdx.x * 256 + threadIdx.x;
    if (idx < (n << shift)) {
        unsigned i = (unsigned)idx >> shift;
        unsigned f = (unsigned)idx & ((1u << shift) - 1u);
        float di = dinv[i];
        float v = z[idx] + di * di * h[idx] + bias[f];
        if (relu) v = fmaxf(v, 0.0f);
        z[idx] = v;
    }
}

// segment max pool (inputs >= 0 after relu, so int-bit atomicMax is exact)
__global__ void k_pool(const float* __restrict__ z, const int* __restrict__ batch,
                       float* __restrict__ pool, int n, int shift)
{
    int idx = blockIdx.x * 256 + threadIdx.x;
    if (idx < (n << shift)) {
        unsigned i = (unsigned)idx >> shift;
        unsigned f = (unsigned)idx & ((1u << shift) - 1u);
        atomicMax((int*)(pool + ((unsigned)batch[i] << shift) + f), __float_as_int(z[idx]));
    }
}

// ---------------------------------------------------------------------------
// Launcher
// ---------------------------------------------------------------------------
extern "C" void kernel_launch(void* const* d_in, const int* in_sizes, int n_in,
                              void* d_out, int out_size, void* d_ws, size_t ws_size,
                              hipStream_t stream)
{
    const float* x_s    = (const float*)d_in[0];
    const float* x_t    = (const float*)d_in[1];
    const int*   ei_s   = (const int*)d_in[2];
    const int*   ei_t   = (const int*)d_in[3];
    const int*   bat_s  = (const int*)d_in[4];
    const int*   bat_t  = (const int*)d_in[5];
    const float* W_enc1 = (const float*)d_in[6];   const float* b_enc1 = (const float*)d_in[7];
    const float* W_enc2 = (const float*)d_in[8];   const float* b_enc2 = (const float*)d_in[9];
    const float* W_r1g1 = (const float*)d_in[10];  const float* b_r1g1 = (const float*)d_in[11];
    const float* W_r1g2 = (const float*)d_in[12];  const float* b_r1g2 = (const float*)d_in[13];
    const float* W_r2g1 = (const float*)d_in[14];  const float* b_r2g1 = (const float*)d_in[15];
    const float* W_r2g2 = (const float*)d_in[16];  const float* b_r2g2 = (const float*)d_in[17];
    const float* W_l1   = (const float*)d_in[18];  const float* b_l1   = (const float*)d_in[19];
    const float* W_l2   = (const float*)d_in[20];  const float* b_l2   = (const float*)d_in[21];

    // ---- workspace carve-up (256B-aligned slabs)
    size_t off = 0;
    auto wsalloc = [&](size_t bytes) -> void* {
        void* p = (char*)d_ws + off;
        off += (bytes + 255) & ~(size_t)255;
        return p;
    };
    bf16_t* bfA    = (bf16_t*)wsalloc((size_t)NN * 2048 * 2);   // activation bf16
    bf16_t* bfW    = (bf16_t*)wsalloc((size_t)2048 * 1024 * 2); // weight bf16 (max 2048x1024)
    float*  hbuf   = (float*) wsalloc((size_t)NN * 2048 * 4);   // GEMM output
    float*  zbuf   = (float*) wsalloc((size_t)NN * 2048 * 4);   // aggregated output
    float*  dinv_s = (float*) wsalloc((size_t)NN * 4);
    float*  dinv_t = (float*) wsalloc((size_t)NN * 4);
    float*  pool_s = (float*) wsalloc((size_t)GG * 2048 * 4);
    float*  pool_t = (float*) wsalloc((size_t)GG * 2048 * 4);
    float*  head1  = (float*) wsalloc((size_t)GG * 1024 * 4);

    auto blocks = [](int n) { return (n + 255) / 256; };
    auto fill = [&](float* p, float v, int n) {   // n always multiple of 4
        k_fill4<<<blocks(n / 4), 256, 0, stream>>>((float4*)p, v, n / 4);
    };
    auto cvt = [&](const float* in, bf16_t* out, int n) {  // n always multiple of 4
        k_cvt_bf16_v4<<<blocks(n / 4), 256, 0, stream>>>((const float4*)in, out, n / 4);
    };
    auto log2i = [](int d) { return d == 512 ? 9 : (d == 1024 ? 10 : 11); };

    // One GCNConv: zbuf = agg(norm * (xin @ W)) + selfloop + bias (+relu)
    auto conv = [&](const float* xin, int n, int din, const float* W, const float* bias,
                    int dout, int relu, const int* src, const int* dst, const float* dinv) {
        cvt(xin, bfA, n * din);
        cvt(W, bfW, din * dout);
        dim3 gg((n + 63) / 64, (dout + 63) / 64);
        k_gemm_bf16<<<gg, 128, 0, stream>>>(bfA, bfW, hbuf, nullptr, n, din, dout, 0);
        fill(zbuf, 0.0f, n * dout);
        dim3 ag(EE, (dout + 1023) / 1024);
        k_edge_agg<<<ag, 256, 0, stream>>>(hbuf, src, dst, dinv, zbuf, dout);
        k_conv_epi<<<blocks(n * dout), 256, 0, stream>>>(zbuf, hbuf, dinv, bias, n, log2i(dout), relu);
    };

    auto run_branch = [&](const float* x, const int* ei, const int* batch,
                          const float* Wenc, const float* benc,
                          const float* Wg1, const float* bg1,
                          const float* Wg2, const float* bg2,
                          float* dinv, float* pool) {
        const int* src = ei;        // edge_index[0]
        const int* dst = ei + EE;   // edge_index[1]
        // degrees incl. self-loops -> dinv = rsqrt(deg)
        fill(dinv, 1.0f, NN);
        k_deg<<<blocks(EE), 256, 0, stream>>>(dst, dinv, EE);
        k_rsqrt<<<blocks(NN), 256, 0, stream>>>(dinv, NN);
        // encoder conv (no relu), then two relu convs
        conv(x,    NN, 128,  Wenc, benc, 512,  0, src, dst, dinv);
        conv(zbuf, NN, 512,  Wg1,  bg1,  1024, 1, src, dst, dinv);
        conv(zbuf, NN, 1024, Wg2,  bg2,  2048, 1, src, dst, dinv);
        // global max pool
        fill(pool, 0.0f, GG * 2048);
        k_pool<<<blocks(NN * 2048), 256, 0, stream>>>(zbuf, batch, pool, NN, 11);
    };

    run_branch(x_s, ei_s, bat_s, W_enc1, b_enc1, W_r1g1, b_r1g1, W_r1g2, b_r1g2, dinv_s, pool_s);
    run_branch(x_t, ei_t, bat_t, W_enc2, b_enc2, W_r2g1, b_r2g1, W_r2g2, b_r2g2, dinv_t, pool_t);

    // pooled sum + MLP head
    k_add<<<blocks(GG * 2048), 256, 0, stream>>>(pool_s, pool_t, GG * 2048);

    cvt(pool_s, bfA, GG * 2048);
    cvt(W_l1, bfW, 2048 * 1024);
    {
        dim3 gg((GG + 63) / 64, (1024 + 63) / 64);
        k_gemm_bf16<<<gg, 128, 0, stream>>>(bfA, bfW, head1, b_l1, GG, 2048, 1024, 1);
    }
    cvt(head1, bfA, GG * 1024);
    cvt(W_l2, bfW, 1024 * 1317);
    float* z_out = (float*)d_out;                 // [G,1317]
    float* s_out = z_out + (size_t)GG * 1317;     // sigmoid(z)
    {
        dim3 gg((GG + 63) / 64, (1317 + 63) / 64);
        k_gemm_bf16<<<gg, 128, 0, stream>>>(bfA, bfW, z_out, b_l2, GG, 1024, 1317, 0);
    }
    k_sigmoid<<<blocks(GG * 1317), 256, 0, stream>>>(z_out, s_out, GG * 1317);
}
